// Attention_23158463660357
// MI455X (gfx1250) — compile-verified, matrix operand order checked by reading, not run on hardware
//
#include <hip/hip_runtime.h>
#include <math.h>

#define T_LEN   2048
#define DMODEL  4096
#define NHEADS  32
#define NGROUPS 8
#define HDIM    128
#define QKV_N   ((NHEADS + 2*NGROUPS)*HDIM)   // 6144

typedef __attribute__((ext_vector_type(16))) __bf16 v16bf;
typedef __attribute__((ext_vector_type(8)))  float  v8f;

struct U4x2 { uint4 a, b; };

static __device__ __forceinline__ v16bf make_frag(uint4 a, uint4 b) {
    U4x2 u{a, b};
    return __builtin_bit_cast(v16bf, u);
}

// round-to-nearest-even f32 -> bf16
static __device__ __forceinline__ unsigned short f2bf(float f) {
    unsigned u = __builtin_bit_cast(unsigned, f);
    u += 0x7FFFu + ((u >> 16) & 1u);
    return (unsigned short)(u >> 16);
}

static __device__ __forceinline__ v8f wmma_bf16(v16bf a, v16bf b, v8f c) {
    return __builtin_amdgcn_wmma_f32_16x16x32_bf16(false, a, false, b,
                                                   (short)0, c, false, false);
}

// Load a 16x32 WMMA operand fragment from LDS.
// Lane layout: row/col = lane&15, k = (lane>>4)*8 + {0..7} and +16.
static __device__ __forceinline__ v16bf lds_frag(const unsigned short* p) {
    uint4 a = *(const uint4*)(p);
    uint4 b = *(const uint4*)(p + 16);
    return make_frag(a, b);
}

#define LSTRIDE 40   // 80B rows: 16B aligned, conflict-free b128 frag reads

// Compute one 128x128 tile of A(f32,MxK) * B(f32,KxN) with on-the-fly bf16
// conversion. 8 waves; wave w owns rows [w*16, w*16+16); acc[j] = cols 16j..16j+15.
static __device__ __forceinline__ void gemm_tile(
    const float* __restrict__ A, const float* __restrict__ B,
    int K, int N, int m0, int n0,
    unsigned short* As, unsigned short* Bs, v8f acc[8])
{
    const int tid  = threadIdx.x;
    const int wave = tid >> 5;
    const int lane = tid & 31;
    const int lrow = lane & 15;
    const int klo  = (lane >> 4) << 3;   // 0 or 8

    for (int j = 0; j < 8; ++j) { v8f z = {}; acc[j] = z; }

    for (int k0 = 0; k0 < K; k0 += 32) {
        __syncthreads();
        // stage A tile 128x32 (f32 -> bf16), row-major
        for (int r = 0; r < 4; ++r) {
            int id  = tid + r*256;            // 0..1023
            int row = id >> 3;                // 0..127
            int c4  = (id & 7) << 2;          // 0..28
            float4 f = *(const float4*)(A + (size_t)(m0+row)*K + k0 + c4);
            unsigned short* dst = As + row*LSTRIDE + c4;
            dst[0]=f2bf(f.x); dst[1]=f2bf(f.y); dst[2]=f2bf(f.z); dst[3]=f2bf(f.w);
        }
        // stage B tile 32x128 (f32 -> bf16), transposed to [n][k]
        for (int r = 0; r < 4; ++r) {
            int id   = tid + r*256;
            int krow = id >> 5;               // 0..31
            int c4   = (id & 31) << 2;        // 0..124
            float4 f = *(const float4*)(B + (size_t)(k0+krow)*N + n0 + c4);
            Bs[(c4+0)*LSTRIDE + krow] = f2bf(f.x);
            Bs[(c4+1)*LSTRIDE + krow] = f2bf(f.y);
            Bs[(c4+2)*LSTRIDE + krow] = f2bf(f.z);
            Bs[(c4+3)*LSTRIDE + krow] = f2bf(f.w);
        }
        __syncthreads();
        v16bf af = lds_frag(As + (wave*16 + lrow)*LSTRIDE + klo);
        for (int j = 0; j < 8; ++j) {
            v16bf bf = lds_frag(Bs + (j*16 + lrow)*LSTRIDE + klo);
            acc[j] = wmma_bf16(af, bf, acc[j]);
        }
    }
}

// ---------------- Kernel 1: QKV projection + RoPE -> bf16 q/k/v ----------------
__global__ __launch_bounds__(256) void qkv_rope_kernel(
    const float* __restrict__ x, const float* __restrict__ w,
    const float* __restrict__ cosT, const float* __restrict__ sinT,
    unsigned short* __restrict__ qbuf,   // [T][NHEADS][HDIM]
    unsigned short* __restrict__ kbuf,   // [T][NGROUPS][HDIM]
    unsigned short* __restrict__ vbuf)   // [T][NGROUPS][HDIM]
{
    __shared__ unsigned short As[128*LSTRIDE];
    __shared__ unsigned short Bs[128*LSTRIDE];
    const int bn = blockIdx.x, bm = blockIdx.y;
    v8f acc[8];
    gemm_tile(x, w, DMODEL, QKV_N, bm*128, bn*128, As, Bs, acc);

    const int tid   = threadIdx.x;
    const int wave  = tid >> 5, lane = tid & 31;
    const int lcol  = lane & 15;
    const int rhalf = lane >> 4;
    const int col0  = bn*128;                  // tile == exactly one head
    const int trow0 = bm*128 + wave*16 + rhalf*8;

    if (col0 < NHEADS*HDIM) {                  // --- Q + RoPE ---
        int head = col0 / HDIM;
        for (int i = 0; i < 8; ++i) {
            int t = trow0 + i;
            const float* cr = cosT + (size_t)t*HDIM;
            const float* sr = sinT + (size_t)t*HDIM;
            unsigned short* orow = qbuf + ((size_t)t*NHEADS + head)*HDIM;
            for (int j = 0; j < 4; ++j) {
                int d1 = j*16 + lcol, d2 = d1 + 64;
                float u1 = acc[j][i], u2 = acc[j+4][i];
                orow[d1] = f2bf(u1*cr[d1] - u2*sr[d1]);
                orow[d2] = f2bf(u2*cr[d2] + u1*sr[d2]);
            }
        }
    } else if (col0 < (NHEADS+NGROUPS)*HDIM) { // --- K + RoPE ---
        int grp = (col0 - NHEADS*HDIM) / HDIM;
        for (int i = 0; i < 8; ++i) {
            int t = trow0 + i;
            const float* cr = cosT + (size_t)t*HDIM;
            const float* sr = sinT + (size_t)t*HDIM;
            unsigned short* orow = kbuf + ((size_t)t*NGROUPS + grp)*HDIM;
            for (int j = 0; j < 4; ++j) {
                int d1 = j*16 + lcol, d2 = d1 + 64;
                float u1 = acc[j][i], u2 = acc[j+4][i];
                orow[d1] = f2bf(u1*cr[d1] - u2*sr[d1]);
                orow[d2] = f2bf(u2*cr[d2] + u1*sr[d2]);
            }
        }
    } else {                                   // --- V ---
        int grp = (col0 - (NHEADS+NGROUPS)*HDIM) / HDIM;
        for (int i = 0; i < 8; ++i) {
            int t = trow0 + i;
            unsigned short* orow = vbuf + ((size_t)t*NGROUPS + grp)*HDIM;
            for (int j = 0; j < 8; ++j)
                orow[j*16 + lcol] = f2bf(acc[j][i]);
        }
    }
}

// ---------------- Kernel 2: causal flash attention ----------------
#define KT        64
#define KS_STRIDE 136   // 272B rows: 16B aligned, conflict-free
#define VT_STRIDE 72    // 144B rows
#define PS_STRIDE 72

__global__ __launch_bounds__(256) void attn_kernel(
    const unsigned short* __restrict__ qbuf,
    const unsigned short* __restrict__ kbuf,
    const unsigned short* __restrict__ vbuf,
    float* __restrict__ attn)                  // [T][NHEADS*HDIM]
{
    __shared__ unsigned short Ks[KT*KS_STRIDE];        // [s][d]
    __shared__ unsigned short Vt[HDIM*VT_STRIDE];      // [d][s] (transposed)
    __shared__ unsigned short Ps[8*16*PS_STRIDE];      // per-wave probs [row][s]

    const int qt    = blockIdx.x;              // query tile (128 rows)
    const int head  = blockIdx.y;
    const int grp   = head >> 2;
    const int tid   = threadIdx.x;
    const int wave  = tid >> 5, lane = tid & 31;
    const int lcol  = lane & 15;
    const int klo   = (lane >> 4) << 3;
    const int rhalf = lane >> 4;
    const int qrow0 = qt*128 + wave*16;

    // Q fragments for this wave's 16 rows (A-operand layout, direct from global)
    v16bf qf[4];
    {
        const unsigned short* qp =
            qbuf + ((size_t)(qrow0 + lcol)*NHEADS + head)*HDIM;
        for (int c = 0; c < 4; ++c) {
            uint4 a = *(const uint4*)(qp + c*32 + klo);
            uint4 b = *(const uint4*)(qp + c*32 + 16 + klo);
            qf[c] = make_frag(a, b);
        }
    }

    v8f oacc[8];
    for (int j = 0; j < 8; ++j) { v8f z = {}; oacc[j] = z; }
    float mrow[8], lrow[8];
    for (int i = 0; i < 8; ++i) { mrow[i] = -3.0e38f; lrow[i] = 0.f; }

    const int   nst   = (qt + 1)*2;            // causal: key tiles of 64
    const float scale = 0.08838834764831845f;  // 1/sqrt(128)
    unsigned short* Pw = Ps + wave*16*PS_STRIDE;

    for (int st = 0; st < nst; ++st) {
        __syncthreads();
        // stage K tile [64][128] and V tile transposed [128][64]
        for (int r = 0; r < 4; ++r) {
            int id   = tid + r*256;            // 0..1023
            int srow = id >> 4;                // 0..63
            int d0   = (id & 15) << 3;         // 0..120
            size_t goff = ((size_t)(st*KT + srow)*NGROUPS + grp)*HDIM + d0;
            *(uint4*)(Ks + srow*KS_STRIDE + d0) = *(const uint4*)(kbuf + goff);
            const unsigned short* vv = vbuf + goff;
            for (int e = 0; e < 8; ++e)
                Vt[(d0+e)*VT_STRIDE + srow] = vv[e];
        }
        __syncthreads();

        if (st*KT <= qrow0 + 15) {             // wave-uniform causal skip
            // scores S = Q K^T
            v8f sacc[4];
            for (int jn = 0; jn < 4; ++jn) {
                v8f s = {};
                const unsigned short* kp = Ks + (jn*16 + lcol)*KS_STRIDE + klo;
                for (int c = 0; c < 4; ++c)
                    s = wmma_bf16(qf[c], lds_frag(kp + c*32), s);
                sacc[jn] = s;
            }
            // scale + causal mask + row max (16-lane half-wave reduction)
            float mnew[8];
            for (int i = 0; i < 8; ++i) {
                int t = qrow0 + rhalf*8 + i;
                float mx = -3.0e38f;
                for (int jn = 0; jn < 4; ++jn) {
                    int s_abs = st*KT + jn*16 + lcol;
                    float v = sacc[jn][i]*scale;
                    if (s_abs > t) v = -3.0e38f;
                    sacc[jn][i] = v;
                    mx = fmaxf(mx, v);
                }
                for (int off = 1; off < 16; off <<= 1)
                    mx = fmaxf(mx, __shfl_xor(mx, off, 32));
                mnew[i] = fmaxf(mrow[i], mx);
            }
            // P = exp(S - m), rescale running state, stash P (bf16) in LDS
            for (int i = 0; i < 8; ++i) {
                float alpha = __expf(mrow[i] - mnew[i]);
                mrow[i] = mnew[i];
                float rs = 0.f;
                int prow = rhalf*8 + i;
                for (int jn = 0; jn < 4; ++jn) {
                    float p = __expf(sacc[jn][i] - mnew[i]);
                    rs += p;
                    Pw[prow*PS_STRIDE + jn*16 + lcol] = f2bf(p);
                }
                for (int off = 1; off < 16; off <<= 1)
                    rs += __shfl_xor(rs, off, 32);
                lrow[i] = lrow[i]*alpha + rs;
                for (int j = 0; j < 8; ++j) oacc[j][i] *= alpha;
            }
            // O += P V
            const unsigned short* pp = Pw + lcol*PS_STRIDE + klo;
            for (int jd = 0; jd < 8; ++jd) {
                const unsigned short* vp = Vt + (jd*16 + lcol)*VT_STRIDE + klo;
                for (int c = 0; c < 2; ++c)
                    oacc[jd] = wmma_bf16(lds_frag(pp + c*32),
                                         lds_frag(vp + c*32), oacc[jd]);
            }
        }
    }
    // normalize and write f32
    for (int i = 0; i < 8; ++i) {
        int t = qrow0 + rhalf*8 + i;
        float inv = 1.0f / lrow[i];
        float* orow = attn + (size_t)t*(NHEADS*HDIM) + head*HDIM;
        for (int jd = 0; jd < 8; ++jd)
            orow[jd*16 + lcol] = oacc[jd][i] * inv;
    }
}

// ---------------- Kernel 3: output projection ----------------
__global__ __launch_bounds__(256) void out_proj_kernel(
    const float* __restrict__ A, const float* __restrict__ W,
    float* __restrict__ out)
{
    __shared__ unsigned short As[128*LSTRIDE];
    __shared__ unsigned short Bs[128*LSTRIDE];
    const int bn = blockIdx.x, bm = blockIdx.y;
    v8f acc[8];
    gemm_tile(A, W, DMODEL, DMODEL, bm*128, bn*128, As, Bs, acc);

    const int tid   = threadIdx.x;
    const int wave  = tid >> 5, lane = tid & 31;
    const int lcol  = lane & 15, rhalf = lane >> 4;
    for (int i = 0; i < 8; ++i) {
        int t = bm*128 + wave*16 + rhalf*8 + i;
        float* orow = out + (size_t)t*DMODEL + bn*128;
        for (int j = 0; j < 8; ++j)
            orow[j*16 + lcol] = acc[j][i];
    }
}

extern "C" void kernel_launch(void* const* d_in, const int* in_sizes, int n_in,
                              void* d_out, int out_size, void* d_ws, size_t ws_size,
                              hipStream_t stream) {
    const float* x     = (const float*)d_in[0];
    const float* w_qkv = (const float*)d_in[1];
    const float* w_out = (const float*)d_in[2];
    const float* cosT  = (const float*)d_in[3];
    const float* sinT  = (const float*)d_in[4];
    // d_in[5] (mask) unused: reference mask is tril -> causal computed analytically

    char* ws = (char*)d_ws;
    unsigned short* qbuf = (unsigned short*)(ws);                          // 16 MB
    unsigned short* kbuf = (unsigned short*)(ws + (size_t)16*1024*1024);   //  4 MB
    unsigned short* vbuf = (unsigned short*)(ws + (size_t)20*1024*1024);   //  4 MB
    float*          attn = (float*)         (ws + (size_t)24*1024*1024);   // 32 MB

    qkv_rope_kernel<<<dim3(QKV_N/128, T_LEN/128), 256, 0, stream>>>(
        x, w_qkv, cosT, sinT, qbuf, kbuf, vbuf);
    attn_kernel<<<dim3(T_LEN/128, NHEADS), 256, 0, stream>>>(
        qbuf, kbuf, vbuf, attn);
    out_proj_kernel<<<dim3(DMODEL/128, T_LEN/128), 256, 0, stream>>>(
        attn, w_out, (float*)d_out);
}
